// InvariantPointAttention_no_z_84464826843876
// MI455X (gfx1250) — matrix operand order repaired and checked by reading, not compiled, not run on hardware
//
#include <hip/hip_runtime.h>
#include <hip/hip_bf16.h>
#include <math.h>
#include <stdint.h>

// ---------------- problem constants ----------------
#define NN    2048
#define CSZ   384
#define NH    12
#define CC    16
#define NPQK  4
#define NPV   8
// augmented logit dim: 16 (qk) + 12 (points) + 2 (q2/k2) + 2 (mask) = 32
#define AD    32
// value dim per head: 16 (v) + 24 (v_pts xyz) + 8 pad = 48
#define VD    48
#define FEATD 576   // H*(C + 4*PV)

typedef __attribute__((ext_vector_type(16))) __bf16        v16bf;
typedef __attribute__((ext_vector_type(8)))  float         v8f;
typedef __attribute__((ext_vector_type(4)))  unsigned int  v4u;
typedef __attribute__((ext_vector_type(8)))  int           v8i;
typedef __attribute__((ext_vector_type(4)))  int           v4i;

struct U64  { unsigned int x, y; };
struct U128 { unsigned int x, y, z, w; };

union BV {
    v16bf          v;
    unsigned short us[16];
    U64            u2[4];
    U128           u4[2];
    __device__ BV() {}
};

static __device__ inline unsigned short f2bfu(float f) {
    unsigned int u = __builtin_bit_cast(unsigned int, f);
    u += 0x7fffu + ((u >> 16) & 1u);     // round-to-nearest-even
    return (unsigned short)(u >> 16);
}

// ---- DPP 16-lane butterfly reductions (stays within each wave half) ----
template <int CTRL>
static __device__ __forceinline__ float dppf(float x) {
    return __builtin_bit_cast(float,
        __builtin_amdgcn_update_dpp(0, __builtin_bit_cast(int, x), CTRL, 0xf, 0xf, true));
}
static __device__ __forceinline__ float redmax16(float x) {
    x = fmaxf(x, dppf<0xB1>(x));    // quad_perm {1,0,3,2}  (xor 1)
    x = fmaxf(x, dppf<0x4E>(x));    // quad_perm {2,3,0,1}  (xor 2)
    x = fmaxf(x, dppf<0x141>(x));   // row_half_mirror      (pairs 4-groups)
    x = fmaxf(x, dppf<0x140>(x));   // row_mirror           (pairs 8-groups)
    return x;
}
static __device__ __forceinline__ float redsum16(float x) {
    x += dppf<0xB1>(x);
    x += dppf<0x4E>(x);
    x += dppf<0x141>(x);
    x += dppf<0x140>(x);
    return x;
}

// A-operand (16x32, 16-bit): lane holds row M=lane&15.
// lanes 0-15: K = 0..15; lanes 16-31: K = 16..31; 2 K per VGPR.
static __device__ inline v16bf load_opA(const unsigned short* row, int hi) {
    BV r; const int k0 = hi * 16;
    r.u2[0] = *(const U64*)(row + k0);
    r.u2[1] = *(const U64*)(row + k0 + 4);
    r.u2[2] = *(const U64*)(row + k0 + 8);
    r.u2[3] = *(const U64*)(row + k0 + 12);
    return r.v;
}

// B-operand (32x16, 16-bit): lane holds col N=lane&15.
// lanes 0-15: K = 0..15; lanes 16-31: K = 16..31 (contiguous -> b128 loads).
static __device__ inline v16bf load_opB(const unsigned short* row, int hi) {
    BV r;
    r.u4[0] = *(const U128*)(row + hi * 16);
    r.u4[1] = *(const U128*)(row + hi * 16 + 8);
    return r.v;
}

// B-operand from a row-major [32][stride] tile: strided u16 gather.
static __device__ inline v16bf load_opB_strided(const unsigned short* base, int stride, int hi) {
    BV r;
#pragma unroll
    for (int e = 0; e < 16; ++e) r.us[e] = base[(hi * 16 + e) * stride];
    return r.v;
}

// ---- Tensor Data Mover: 1-D contiguous copy global -> LDS ----
// D# group0/group1 per CDNA5 ISA ch.8 (data_size=1B, 1-row tile of nbytes).
// clang-23 toolchain: 6-arg builtin (g0 u32x4, g1 i32x8, g2 i32x4, g3 i32x4, i32x8, cpol)
#if __has_builtin(__builtin_amdgcn_tensor_load_to_lds)
#define HAVE_TDM 1
static __device__ inline void tdm_load_1d(unsigned int lds_addr, const void* gptr,
                                          unsigned int nbytes) {
    unsigned long long ga = (unsigned long long)(uintptr_t)gptr;
    v4u g0;
    g0[0] = 1u;                                      // count=1 (valid), user mode
    g0[1] = lds_addr;                                // lds_addr (bytes)
    g0[2] = (unsigned int)ga;                        // global_addr[31:0]
    g0[3] = (unsigned int)((ga >> 32) & 0x01ffffffu) // global_addr[56:32]
            | 0x80000000u;                           // type=2 ("image")
    v8i g1;
    g1[0] = 0;                                       // wg_mask=0, data_size=1B
    g1[1] = (int)(nbytes << 16);                     // tensor_dim0[15:0]
    g1[2] = (int)(((nbytes >> 16) & 0xffffu) | (1u << 16)); // dim0 hi | tensor_dim1=1
    g1[3] = (int)(nbytes << 16);                     // tile_dim0 = nbytes (<=64KB)
    g1[4] = 1;                                       // tile_dim1=1
    g1[5] = (int)nbytes;                             // tensor_dim0_stride lo
    g1[6] = 0;
    g1[7] = 0;
    v4i gz  = {0, 0, 0, 0};
    v8i gz8 = {0, 0, 0, 0, 0, 0, 0, 0};
    __builtin_amdgcn_tensor_load_to_lds(g0, g1, gz, gz, gz8, 0);
}
static __device__ inline void tdm_wait0() { __builtin_amdgcn_s_wait_tensorcnt(0); }
#else
#define HAVE_TDM 0
#endif

// =====================================================================
// Kernel 1: projections + frame transform + build augmented Q^,K^,V^ (bf16)
// one block per residue n
// =====================================================================
__global__ __launch_bounds__(256) void k_proj(
    const float* __restrict__ s,   const float* __restrict__ rot,
    const float* __restrict__ trans, const float* __restrict__ mask,
    const float* __restrict__ Wq,  const float* __restrict__ bq,
    const float* __restrict__ Wkv, const float* __restrict__ bkv,
    const float* __restrict__ Wqp, const float* __restrict__ bqp,
    const float* __restrict__ Wkvp,const float* __restrict__ bkvp,
    const float* __restrict__ head_weights,
    unsigned short* __restrict__ Qh, unsigned short* __restrict__ Kh,
    unsigned short* __restrict__ Vh)
{
    __shared__ __attribute__((aligned(16))) float s_s[CSZ];
    __shared__ float raw[1152];          // q 192 | kv 384 | qp 144 | kvp 432
    __shared__ float qpt[48][3];
    __shared__ float kvpt[144][3];
    __shared__ float q2[NH], k2[NH], hw[NH];
    __shared__ float Rsh[9], Tsh[3];

    const int n = blockIdx.x, t = threadIdx.x;

    for (int i = t; i < CSZ; i += 256) s_s[i] = s[n * CSZ + i];
    if (t < NH) {
        float w  = head_weights[t];
        float sp = (w > 20.f) ? w : log1pf(__expf(w));       // softplus
        hw[t] = sp * sqrtf(1.0f / 54.0f);                    // sqrt(1/(3*PQK*9/2))
        q2[t] = 0.f; k2[t] = 0.f;
    }
    if (t < 9) Rsh[t] = rot[n * 9 + t];
    if (t < 3) Tsh[t] = trans[n * 3 + t];
    __syncthreads();

    // all projections: 1152 dot products of length 384 (float4 vectorized)
    for (int o = t; o < 1152; o += 256) {
        const float* Wrow; float bias;
        if      (o < 192)  { Wrow = Wq   + o * CSZ;          bias = bq[o];        }
        else if (o < 576)  { int r = o - 192; Wrow = Wkv  + r * CSZ; bias = bkv[r];  }
        else if (o < 720)  { int r = o - 576; Wrow = Wqp  + r * CSZ; bias = bqp[r];  }
        else               { int r = o - 720; Wrow = Wkvp + r * CSZ; bias = bkvp[r]; }
        const float4* W4 = (const float4*)Wrow;
        const float4* S4 = (const float4*)s_s;
        float acc = bias;
        for (int k = 0; k < CSZ / 4; ++k) {
            float4 w = W4[k], x = S4[k];
            acc = fmaf(x.x, w.x, acc);
            acc = fmaf(x.y, w.y, acc);
            acc = fmaf(x.z, w.z, acc);
            acc = fmaf(x.w, w.w, acc);
        }
        raw[o] = acc;
    }
    __syncthreads();

    // transform points to global frame; accumulate q2/k2 via LDS float atomics
    for (int idx = t; idx < 192; idx += 256) {
        float lx, ly, lz;
        if (idx < 48) { lx = raw[576 + idx]; ly = raw[624 + idx]; lz = raw[672 + idx]; }
        else { int m = idx - 48; lx = raw[720 + m]; ly = raw[864 + m]; lz = raw[1008 + m]; }
        float gx = Rsh[0]*lx + Rsh[1]*ly + Rsh[2]*lz + Tsh[0];
        float gy = Rsh[3]*lx + Rsh[4]*ly + Rsh[5]*lz + Tsh[1];
        float gz = Rsh[6]*lx + Rsh[7]*ly + Rsh[8]*lz + Tsh[2];
        float sq = gx*gx + gy*gy + gz*gz;
        if (idx < 48) {
            qpt[idx][0] = gx; qpt[idx][1] = gy; qpt[idx][2] = gz;
            atomicAdd(&q2[idx / NPQK], sq);
        } else {
            int m = idx - 48;
            kvpt[m][0] = gx; kvpt[m][1] = gy; kvpt[m][2] = gz;
            if ((m % 12) < NPQK) atomicAdd(&k2[m / 12], sq);
        }
    }
    __syncthreads();

    const float mi = mask[n];
    // augmented Q^ / K^ vectors (bf16), layout [h][n][32]
    for (int e = t; e < NH * AD * 2; e += 256) {
        const int h = e >> 6, d = (e >> 1) & 31, isK = e & 1;
        const float shw = sqrtf(hw[h]);
        float val;
        if (!isK) {
            if      (d < 16) val = raw[h * 16 + d] * 0.14433757f;   // sqrt(1/(3C))
            else if (d < 28) { int pd = d - 16; val = qpt[h*4 + pd/3][pd%3] * shw; }
            else if (d == 28) val = -0.5f * hw[h] * q2[h];
            else if (d == 29) val = 1.0f;
            else if (d == 30) val = 316.22776f * mi;                // sqrt(INF)*m_i
            else              val = -100000.0f;                     // -INF
        } else {
            if      (d < 16) val = raw[192 + h * 32 + d];
            else if (d < 28) { int pd = d - 16; val = kvpt[h*12 + pd/3][pd%3] * shw; }
            else if (d == 28) val = 1.0f;
            else if (d == 29) val = -0.5f * hw[h] * k2[h];
            else if (d == 30) val = 316.22776f * mi;                // sqrt(INF)*m_j
            else              val = 1.0f;
        }
        unsigned short b = f2bfu(val);
        if (!isK) Qh[(h * NN + n) * AD + d] = b;
        else      Kh[(h * NN + n) * AD + d] = b;
    }
    // augmented V^: [h][n][48] = v(16) | v_pts xyz interleaved (24) | pad(8)
    for (int e = t; e < NH * VD; e += 256) {
        const int h = e / VD, d = e % VD;
        float val;
        if      (d < 16) val = raw[192 + h * 32 + 16 + d];
        else if (d < 40) { int pd = d - 16; val = kvpt[h*12 + 4 + pd/3][pd%3]; }
        else             val = 0.0f;
        Vh[(h * NN + n) * VD + d] = f2bfu(val);
    }
}

// =====================================================================
// Kernel 2: flash attention. grid = (N/128, H), 256 threads = 8 waves,
// each wave owns one 16-row i-tile; K/V tiles DMA'd to LDS by the TDM.
// =====================================================================
__global__ __launch_bounds__(256) void k_attn(
    const unsigned short* __restrict__ Qh, const unsigned short* __restrict__ Kh,
    const unsigned short* __restrict__ Vh,
    const float* __restrict__ rot, const float* __restrict__ trans,
    float* __restrict__ feats)
{
    __shared__ __attribute__((aligned(16))) unsigned short q_s[128 * AD];
    __shared__ __attribute__((aligned(16))) unsigned short k_s[32 * AD];
    __shared__ __attribute__((aligned(16))) unsigned short v_s[32 * VD];   // raw rows
    __shared__ __attribute__((aligned(16))) unsigned short p_s[8][16 * 32];
    __shared__ float o_s[8][16 * VD];

    const int h = blockIdx.y, i0 = blockIdx.x * 128;
    const int t = threadIdx.x, wave = t >> 5, lane = t & 31;
    const int hi = lane >> 4, ln = lane & 15;

#if HAVE_TDM
    const unsigned int q_lds = (unsigned int)(uintptr_t)q_s;
    const unsigned int k_lds = (unsigned int)(uintptr_t)k_s;
    const unsigned int v_lds = (unsigned int)(uintptr_t)v_s;
    if (wave == 0) {
        tdm_load_1d(q_lds, Qh + (size_t)(h * NN + i0) * AD, 128 * AD * 2);
        tdm_wait0();
    }
#else
    {
        const unsigned int* qg = (const unsigned int*)(Qh + (size_t)(h * NN + i0) * AD);
        unsigned int* ql = (unsigned int*)q_s;
        for (int i = t; i < 128 * AD / 2; i += 256) ql[i] = qg[i];
    }
#endif
    __syncthreads();

    const v16bf aq = load_opA(q_s + (wave * 16 + ln) * AD, hi);

    v8f acc0 = {}, acc1 = {}, acc2 = {};
    float m_[8], l_[8];
#pragma unroll
    for (int r = 0; r < 8; ++r) { m_[r] = -30000.f; l_[r] = 0.f; }

    for (int jb = 0; jb < NN; jb += 32) {
        __syncthreads();
#if HAVE_TDM
        if (wave == 0) {
            tdm_load_1d(k_lds, Kh + (size_t)(h * NN + jb) * AD, 32 * AD * 2);
            tdm_load_1d(v_lds, Vh + (size_t)(h * NN + jb) * VD, 32 * VD * 2);
            tdm_wait0();
        }
#else
        {
            const unsigned int* kg = (const unsigned int*)(Kh + (size_t)(h * NN + jb) * AD);
            unsigned int* kl = (unsigned int*)k_s;
            for (int i = t; i < 32 * AD / 2; i += 256) kl[i] = kg[i];
            const unsigned int* vg = (const unsigned int*)(Vh + (size_t)(h * NN + jb) * VD);
            unsigned int* vl = (unsigned int*)v_s;
            for (int i = t; i < 32 * VD / 2; i += 256) vl[i] = vg[i];
        }
#endif
        if (jb + 32 < NN) {     // gfx1250 global_prefetch of next tiles into GL2
            __builtin_prefetch(Kh + (size_t)(h * NN + jb + 32) * AD, 0, 1);
            __builtin_prefetch(Vh + (size_t)(h * NN + jb + 32) * VD, 0, 1);
        }
        __syncthreads();

        // S = Q^ . K^T : two 16x16 logit tiles via bf16 WMMA
        v16bf b0 = load_opB(k_s + ln * AD, hi);
        v16bf b1 = load_opB(k_s + (16 + ln) * AD, hi);
        v8f S0 = {}, S1 = {};
        S0 = __builtin_amdgcn_wmma_f32_16x16x32_bf16(false, aq, false, b0, (short)0, S0, false, false);
        S1 = __builtin_amdgcn_wmma_f32_16x16x32_bf16(false, aq, false, b1, (short)0, S1, false, false);

        // online softmax: accumulator row r maps to matrix row (r + 8*hi);
        // per-row stats via DPP butterflies (no LDS traffic)
        unsigned short* pw = p_s[wave];
#pragma unroll
        for (int r = 0; r < 8; ++r) {
            float s0 = S0[r], s1 = S1[r];
            float mx = redmax16(fmaxf(s0, s1));
            float mn = fmaxf(m_[r], mx);
            float sc = __expf(m_[r] - mn);
            float p0 = __expf(s0 - mn), p1 = __expf(s1 - mn);
            float rs = redsum16(p0 + p1);
            l_[r] = l_[r] * sc + rs;
            m_[r] = mn;
            acc0[r] *= sc; acc1[r] *= sc; acc2[r] *= sc;
            int row = r + hi * 8;
            pw[row * 32 + ln]      = f2bfu(p0);
            pw[row * 32 + 16 + ln] = f2bfu(p1);
        }
        __syncthreads();

        // O += P . V^ (48 cols -> 3 WMMAs); V rows are raw [j][48] in LDS
        v16bf ap  = load_opA(pw + ln * 32, hi);
        v16bf vb0 = load_opB_strided(v_s + ln,      VD, hi);
        v16bf vb1 = load_opB_strided(v_s + 16 + ln, VD, hi);
        v16bf vb2 = load_opB_strided(v_s + 32 + ln, VD, hi);
        acc0 = __builtin_amdgcn_wmma_f32_16x16x32_bf16(false, ap, false, vb0, (short)0, acc0, false, false);
        acc1 = __builtin_amdgcn_wmma_f32_16x16x32_bf16(false, ap, false, vb1, (short)0, acc1, false, false);
        acc2 = __builtin_amdgcn_wmma_f32_16x16x32_bf16(false, ap, false, vb2, (short)0, acc2, false, false);
    }

    // normalize and park in LDS for the row-wise epilogue
    {
        float* ow = o_s[wave];
#pragma unroll
        for (int r = 0; r < 8; ++r) {
            float inv = 1.0f / l_[r];
            int row = r + hi * 8;
            ow[row * VD + ln]      = acc0[r] * inv;
            ow[row * VD + 16 + ln] = acc1[r] * inv;
            ow[row * VD + 32 + ln] = acc2[r] * inv;
        }
    }
    __syncthreads();

    // epilogue: inverse frame transform + dists, scatter into feats[N][576]
    if (lane < 16) {
        const int i = i0 + wave * 16 + lane;
        const float* o = o_s[wave] + lane * VD;
        float R[9], T[3];
#pragma unroll
        for (int e = 0; e < 9; ++e) R[e] = rot[i * 9 + e];
#pragma unroll
        for (int e = 0; e < 3; ++e) T[e] = trans[i * 3 + e];
        float* f = feats + (size_t)i * FEATD;
#pragma unroll
        for (int c = 0; c < CC; ++c) f[h * CC + c] = o[c];
#pragma unroll
        for (int p = 0; p < NPV; ++p) {
            float gx = o[16 + p * 3 + 0] - T[0];
            float gy = o[16 + p * 3 + 1] - T[1];
            float gz = o[16 + p * 3 + 2] - T[2];
            float lx = R[0]*gx + R[3]*gy + R[6]*gz;   // R^T (g - t)
            float ly = R[1]*gx + R[4]*gy + R[7]*gz;
            float lz = R[2]*gx + R[5]*gy + R[8]*gz;
            float d  = sqrtf(lx*lx + ly*ly + lz*lz + 1e-8f);
            int idx = h * NPV + p;
            f[192 + idx] = lx; f[288 + idx] = ly; f[384 + idx] = lz; f[480 + idx] = d;
        }
    }
}

// =====================================================================
// Kernel 3: out = feats @ Wout^T + bout  (2048 x 384, K=576), bf16 WMMA.
// One wave per 16x16 output tile; operands converted from f32 on the fly.
// =====================================================================
static __device__ inline void pack4(unsigned short* us, const float4 f) {
    us[0] = f2bfu(f.x); us[1] = f2bfu(f.y); us[2] = f2bfu(f.z); us[3] = f2bfu(f.w);
}

__global__ __launch_bounds__(256) void k_out(
    const float* __restrict__ feats, const float* __restrict__ Wout,
    const float* __restrict__ bout, float* __restrict__ out)
{
    const int wave = threadIdx.x >> 5, lane = threadIdx.x & 31;
    const int tile = blockIdx.x * 8 + wave;
    const int ti = tile / (CSZ / 16), tj = tile % (CSZ / 16);
    const int hi = lane >> 4, ln = lane & 15;

    const float4* pa = (const float4*)(feats + (size_t)(ti * 16 + ln) * FEATD);
    const float4* pb = (const float4*)(Wout  + (size_t)(tj * 16 + ln) * FEATD);

    v8f acc = {};
    for (int kb = 0; kb < FEATD; kb += 32) {
        BV av, bv;
        const int ka = (kb + hi * 16) >> 2;       // A: K = kb + hi*16 + {0..15}
        pack4(av.us + 0,  pa[ka]);
        pack4(av.us + 4,  pa[ka + 1]);
        pack4(av.us + 8,  pa[ka + 2]);
        pack4(av.us + 12, pa[ka + 3]);
        const int kbq = (kb + hi * 16) >> 2;      // B: K = kb + hi*16 + {0..15}
        pack4(bv.us + 0,  pb[kbq]);
        pack4(bv.us + 4,  pb[kbq + 1]);
        pack4(bv.us + 8,  pb[kbq + 2]);
        pack4(bv.us + 12, pb[kbq + 3]);
        acc = __builtin_amdgcn_wmma_f32_16x16x32_bf16(false, av.v, false, bv.v, (short)0, acc, false, false);
    }

    const int ocol = tj * 16 + ln;
    const float bias = bout[ocol];
#pragma unroll
    for (int r = 0; r < 8; ++r) {
        int i = ti * 16 + r + hi * 8;
        out[i * CSZ + ocol] = acc[r] + bias;
    }
}

// =====================================================================
extern "C" void kernel_launch(void* const* d_in, const int* in_sizes, int n_in,
                              void* d_out, int out_size, void* d_ws, size_t ws_size,
                              hipStream_t stream) {
    const float* s     = (const float*)d_in[0];
    const float* rot   = (const float*)d_in[1];
    const float* trans = (const float*)d_in[2];
    const float* mask  = (const float*)d_in[3];
    const float* Wq    = (const float*)d_in[4];
    const float* bq    = (const float*)d_in[5];
    const float* Wkv   = (const float*)d_in[6];
    const float* bkv   = (const float*)d_in[7];
    const float* Wqp   = (const float*)d_in[8];
    const float* bqp   = (const float*)d_in[9];
    const float* Wkvp  = (const float*)d_in[10];
    const float* bkvp  = (const float*)d_in[11];
    const float* hwts  = (const float*)d_in[12];
    const float* Wout  = (const float*)d_in[13];
    const float* bout  = (const float*)d_in[14];
    float* out = (float*)d_out;

    unsigned short* Qh = (unsigned short*)d_ws;            // H*N*32 bf16
    unsigned short* Kh = Qh + (size_t)NH * NN * AD;        // H*N*32 bf16
    unsigned short* Vh = Kh + (size_t)NH * NN * AD;        // H*N*48 bf16
    float* feats = (float*)(Vh + (size_t)NH * NN * VD);    // N*576 f32

    k_proj<<<NN, 256, 0, stream>>>(s, rot, trans, mask, Wq, bq, Wkv, bkv,
                                   Wqp, bqp, Wkvp, bkvp, hwts, Qh, Kh, Vh);
    k_attn<<<dim3(NN / 128, NH), 256, 0, stream>>>(Qh, Kh, Vh, rot, trans, feats);
    k_out<<<(NN / 16) * (CSZ / 16) / 8, 256, 0, stream>>>(feats, Wout, bout, out);
}
